// ESRNN_42262478193329
// MI455X (gfx1250) — matrix operand order repaired and checked by reading, not compiled, not run on hardware
//
#include <hip/hip_runtime.h>

// ---------------------------------------------------------------------------
// ES-RNN forward for MI455X (gfx1250, wave32).
// bf16 activations + f32 accumulate via v_wmma_f32_16x16x32_bf16.
// Fused dilated-LSTM: gates = x@Wih^T + h@Whh^T + bias per step, all on WMMA,
// with 4 independent accumulator chains per wave (fills WMMA hazard slots).
// ---------------------------------------------------------------------------

typedef __attribute__((ext_vector_type(16))) __bf16 v16bf;
typedef __attribute__((ext_vector_type(8)))  __bf16 v8bf;
typedef __attribute__((ext_vector_type(8)))  float  v8f;

#define NS    2048      // series
#define TT    480       // train length
#define NW    451       // windows = 480 - 18 - 11
#define HID   50
#define NG    200       // 4*HID gates
#define NGP   208       // padded to 13 N-tiles of 16
#define TPM   456       // max padded time (d=12)
#define KXMAX 96        // max padded input width (layer 3: 68 -> 96)

__device__ __forceinline__ float sigm(float x){ return 1.f/(1.f+__expf(-x)); }

#define WMMA_BF16(A,B,C) __builtin_amdgcn_wmma_f32_16x16x32_bf16(false,(A),false,(B),(short)0,(C),false,false)

// ---- WMMA bf16 operand loaders (row-major source, stride in elements) ------
// A 16x32: lane l holds row (l&15); elems 0..15 -> k = 16*(l>>4)+{0..15}
//          (two 16B loads)
__device__ __forceinline__ v16bf load_a_op(const __bf16* base, int stride,
                                           int row0, int k0, int lane){
  int m  = row0 + (lane & 15);
  int kh = (lane >> 4) & 1;
  const __bf16* p = base + (size_t)m * stride + k0 + kh * 16;
  v8bf lo = *(const v8bf*)p;
  v8bf hi = *(const v8bf*)(p + 8);
  v16bf r;
#pragma unroll
  for (int i = 0; i < 8; ++i){ r[i] = lo[i]; r[i+8] = hi[i]; }
  return r;
}
// B 32x16 (= rows of W, since gates = h @ W^T): lane l holds W row (n0+(l&15)),
// 16 consecutive k starting at k0 + 16*(l>>4)  (one 32B load)
__device__ __forceinline__ v16bf load_b_op(const __bf16* W, int stride,
                                           int n0, int k0, int lane){
  int n  = n0 + (lane & 15);
  int kh = (lane >> 4) & 1;
  return *(const v16bf*)(W + (size_t)n * stride + k0 + kh * 16);
}

// ---------------------------------------------------------------------------
__global__ void zero_ws_kernel(uint4* p, size_t n16){
  size_t i = (size_t)blockIdx.x * blockDim.x + threadIdx.x;
  size_t stride = (size_t)gridDim.x * blockDim.x;
  uint4 z; z.x = z.y = z.z = z.w = 0u;
  for (; i < n16; i += stride) p[i] = z;
}

__global__ void conv_weight_kernel(const float* src, __bf16* dst,
                                   int N, int K, int Np, int Kp){
  int i = blockIdx.x * blockDim.x + threadIdx.x;
  if (i >= Np * Kp) return;
  int n = i / Kp, k = i % Kp;
  dst[i] = (n < N && k < K) ? (__bf16)src[n * K + k] : (__bf16)0.f;
}

__global__ void bias_sum_kernel(const float* b1, const float* b2, float* o, int n){
  int i = blockIdx.x * blockDim.x + threadIdx.x;
  if (i < n) o[i] = b1[i] + b2[i];
}

// ---- Holt-Winters smoothing: one lane per series ---------------------------
__global__ void smoothing_kernel(const float* train, const float* ilev,
                                 const float* isea, const float* iseason,
                                 float* levs, float* seas){
  int s = blockIdx.x * blockDim.x + threadIdx.x;
  if (s >= NS) return;
  float lsm = sigm(ilev[s]);
  float ssm = sigm(isea[s]);
  float buf[12];
  float S0 = __expf(iseason[s*12 + 0]);
  seas[(size_t)s*492 + 0] = S0;
#pragma unroll
  for (int k = 1; k < 12; ++k){
    float v = __expf(iseason[s*12 + k]);
    seas[(size_t)s*492 + k] = v;
    buf[k-1] = v;
  }
  buf[11] = S0;
  seas[(size_t)s*492 + 12] = S0;
  float lev = train[(size_t)s*TT] / S0;
  levs[(size_t)s*TT] = lev;
  int head = 0;
  for (int t = 1; t < TT; ++t){
    float y  = train[(size_t)s*TT + t];
    float si = buf[head];
    float nl = lsm * (y / si) + (1.f - lsm) * lev;
    float nsv = ssm * (y / nl) + (1.f - ssm) * si;
    buf[head] = nsv;                 // pop front, push back
    head = (head + 1) % 12;
    lev = nl;
    levs[(size_t)s*TT + t]       = nl;
    seas[(size_t)s*492 + t + 12] = nsv;
  }
}

// ---- window build: input (bf16, K pad 32) + output_batch (f32 -> d_out) ----
__global__ void build_input_kernel(const float* train, const float* cat,
                                   const float* levs, const float* seas,
                                   __bf16* inp, float* outB){
  int id = blockIdx.x * blockDim.x + threadIdx.x;
  if (id >= NW * NS) return;
  int w = id / NS, b = id % NS;
  int i = 11 + w;
  float lev = levs[(size_t)b*TT + i];
  size_t r = (size_t)w * NS + b;
#pragma unroll
  for (int j = 0; j < 12; ++j){
    float x = train[(size_t)b*TT + w + j] / seas[(size_t)b*492 + w + j] / lev;
    inp[r*32 + j] = (__bf16)x;
  }
#pragma unroll
  for (int j = 0; j < 6; ++j)
    inp[r*32 + 12 + j] = (__bf16)cat[(size_t)b*6 + j];
#pragma unroll
  for (int k = 0; k < 18; ++k){
    int t = i + 1 + k;
    outB[r*18 + k] = train[(size_t)b*TT + t] / seas[(size_t)b*492 + t] / lev;
  }
}

// ---- head GEMM: C[M][N] = A[M][Kp] * W[N][Kp]^T + b ------------------------
// One wave computes 16 rows x NTC N-tiles (compile-time), reusing each A
// operand NTC times; NTC independent accumulator chains, branchless K-loop.
template<int NTC>
__global__ void gemm_nt4_kernel(const __bf16* A, int lda,
                                const __bf16* W, int ldw, const float* bias,
                                int M, int N, int Kp,
                                float* outF, int ldcf,
                                __bf16* outH, int ldch, int act){
  int lane = threadIdx.x;            // 32 threads = 1 wave
  int m0 = blockIdx.x * 16;
  v8f acc[NTC];
#pragma unroll
  for (int q = 0; q < NTC; ++q) acc[q] = (v8f){};
  for (int k0 = 0; k0 < Kp; k0 += 32){
    v16bf a = load_a_op(A, lda, m0, k0, lane);
#pragma unroll
    for (int q = 0; q < NTC; ++q)
      acc[q] = WMMA_BF16(a, load_b_op(W, ldw, q * 16, k0, lane), acc[q]);
  }
  int hi = lane >> 4;
#pragma unroll
  for (int q = 0; q < NTC; ++q){
    int n = q * 16 + (lane & 15);
    float bv = (bias && n < N) ? bias[n] : 0.f;
#pragma unroll
    for (int r = 0; r < 8; ++r){
      int m = m0 + r + 8 * hi;
      if (n < N && m < M){
        float v = acc[q][r] + bv;
        if (act) v = tanhf(v);
        if (outF) outF[(size_t)m * ldcf + n] = v;
        if (outH) outH[(size_t)m * ldch + n] = (__bf16)v;
      }
    }
  }
}

// ---- fully fused dilated-LSTM ----------------------------------------------
// One workgroup = 16 batch rows; 4 waves x 4 N-tiles cover the 13 gate tiles
// (waves 1-3 compute a harmless duplicate of tile 12 as their 4th tile and
// skip its store -> hot loop is completely branchless).
// Per step: gates = x@Wih^T (Wih in LDS, x streamed bf16 from HBM)
//                 + h@Whh^T (Whh B-operands resident in registers)
//                 + (bih + bhh).
// h (bf16) and c (f32) stay in LDS for the whole time loop.
__device__ __forceinline__ void store_gate_tile(const v8f& acc, int nt, bool ok,
                                                int lane, const float* bias_lds,
                                                float* gates){
  int n  = nt * 16 + (lane & 15);
  int hi = lane >> 4;
  if (ok && n < NG){
#pragma unroll
    for (int r = 0; r < 8; ++r)
      gates[(r + 8*hi) * NGP + n] = acc[r] + bias_lds[n];
  }
}

template<int KXC>                                    // k-chunks of x: Kxp = 32*KXC
__global__ void __launch_bounds__(128)
lstm_fused_kernel(const __bf16* X,                   // layer input [steps*Bd][KXP]
                  const __bf16* WihG,                // [208][KXP] bf16
                  const __bf16* Whh,                 // [208][64]  bf16
                  const float* bsum,                 // bih + bhh  [200]
                  __bf16* hOut, int steps, int Bd){
  constexpr int KXP = KXC * 32;
  __shared__ __bf16 wih_lds[NGP * KXP];
  __shared__ __bf16 h_tile[16 * 64];
  __shared__ float  c_tile[16 * HID];
  __shared__ float  gates[16 * NGP];
  __shared__ float  bias_lds[NGP];
  int tid = threadIdx.x, lane = tid & 31, wave = tid >> 5;
  int row0 = blockIdx.x * 16;

  // stage Wih into LDS (16B vectors), biases, zero h/c state
  for (int i = tid; i < (NGP * KXP) >> 3; i += 128)
    ((uint4*)wih_lds)[i] = ((const uint4*)WihG)[i];
  for (int i = tid; i < NGP; i += 128) bias_lds[i] = (i < NG) ? bsum[i] : 0.f;
  for (int i = tid; i < 16 * 64;  i += 128) h_tile[i] = (__bf16)0.f;
  for (int i = tid; i < 16 * HID; i += 128) c_tile[i] = 0.f;

  // tile assignment: wave w owns tiles w, w+4, w+8, w+12; the 4th is clamped
  // to tile 12 for waves 1-3 (duplicate compute, store suppressed)
  int nt0 = wave, nt1 = wave + 4, nt2 = wave + 8, nt3 = wave + 12;
  bool t3ok = (nt3 < 13);
  int nt3c = t3ok ? nt3 : 12;

  // Whh B-operands resident in registers
  v16bf b00 = load_b_op(Whh, 64, nt0*16, 0,  lane);
  v16bf b01 = load_b_op(Whh, 64, nt0*16, 32, lane);
  v16bf b10 = load_b_op(Whh, 64, nt1*16, 0,  lane);
  v16bf b11 = load_b_op(Whh, 64, nt1*16, 32, lane);
  v16bf b20 = load_b_op(Whh, 64, nt2*16, 0,  lane);
  v16bf b21 = load_b_op(Whh, 64, nt2*16, 32, lane);
  v16bf b30 = load_b_op(Whh, 64, nt3c*16, 0,  lane);
  v16bf b31 = load_b_op(Whh, 64, nt3c*16, 32, lane);
  __syncthreads();

  for (int t = 0; t < steps; ++t){
    int grow = t * Bd + row0;
    if (t + 1 < steps)
      __builtin_prefetch(X + ((size_t)(t+1) * Bd + row0) * KXP, 0, 1);

    // 4 independent accumulator chains, advanced together per k-chunk
    v8f g0 = {}, g1 = {}, g2 = {}, g3 = {};
#pragma unroll
    for (int c = 0; c < KXC; ++c){
      v16bf ax = load_a_op(X, KXP, grow, c * 32, lane);
      g0 = WMMA_BF16(ax, load_b_op(wih_lds, KXP, nt0*16,  c*32, lane), g0);
      g1 = WMMA_BF16(ax, load_b_op(wih_lds, KXP, nt1*16,  c*32, lane), g1);
      g2 = WMMA_BF16(ax, load_b_op(wih_lds, KXP, nt2*16,  c*32, lane), g2);
      g3 = WMMA_BF16(ax, load_b_op(wih_lds, KXP, nt3c*16, c*32, lane), g3);
    }
    v16bf ah0 = load_a_op(h_tile, 64, 0, 0, lane);
    g0 = WMMA_BF16(ah0, b00, g0);
    g1 = WMMA_BF16(ah0, b10, g1);
    g2 = WMMA_BF16(ah0, b20, g2);
    g3 = WMMA_BF16(ah0, b30, g3);
    v16bf ah1 = load_a_op(h_tile, 64, 0, 32, lane);
    g0 = WMMA_BF16(ah1, b01, g0);
    g1 = WMMA_BF16(ah1, b11, g1);
    g2 = WMMA_BF16(ah1, b21, g2);
    g3 = WMMA_BF16(ah1, b31, g3);

    store_gate_tile(g0, nt0, true, lane, bias_lds, gates);
    store_gate_tile(g1, nt1, true, lane, bias_lds, gates);
    store_gate_tile(g2, nt2, true, lane, bias_lds, gates);
    store_gate_tile(g3, nt3, t3ok, lane, bias_lds, gates);
    __syncthreads();

    for (int idx = tid; idx < 16 * HID; idx += 128){
      int lb = idx / HID, hh = idx % HID;
      float gi = gates[lb * NGP + hh];
      float gf = gates[lb * NGP + hh + HID];
      float gg = gates[lb * NGP + hh + 2 * HID];
      float go = gates[lb * NGP + hh + 3 * HID];
      float c  = sigm(gf) * c_tile[idx] + sigm(gi) * tanhf(gg);
      float h  = sigm(go) * tanhf(c);
      c_tile[idx] = c;
      __bf16 hb = (__bf16)h;
      h_tile[lb * 64 + hh] = hb;
      hOut[((size_t)t * Bd + row0 + lb) * 64 + hh] = hb;
    }
    __syncthreads();
  }
}

// ---- concat [input(18) | h2(50)] -> 96-padded bf16 -------------------------
__global__ void concat_kernel(const __bf16* inp, const __bf16* h2, __bf16* cc){
  int id = blockIdx.x * blockDim.x + threadIdx.x;
  if (id >= NW * NS) return;
  size_t r = (size_t)id;
#pragma unroll
  for (int j = 0; j < 18; ++j) cc[r*96 + j] = inp[r*32 + j];
#pragma unroll
  for (int j = 0; j < HID; ++j) cc[r*96 + 18 + j] = h2[r*64 + j];
}

// ---------------------------------------------------------------------------
extern "C" void kernel_launch(void* const* d_in, const int* in_sizes, int n_in,
                              void* d_out, int out_size, void* d_ws, size_t ws_size,
                              hipStream_t stream){
  (void)in_sizes; (void)n_in; (void)out_size; (void)ws_size;

  const float* train    = (const float*)d_in[0];
  const float* info_cat = (const float*)d_in[3];
  const float* ilev     = (const float*)d_in[5];
  const float* isea     = (const float*)d_in[6];
  const float* iseason  = (const float*)d_in[7];
  const float* Wih[4] = {(const float*)d_in[8],  (const float*)d_in[12],
                         (const float*)d_in[16], (const float*)d_in[20]};
  const float* Whh[4] = {(const float*)d_in[9],  (const float*)d_in[13],
                         (const float*)d_in[17], (const float*)d_in[21]};
  const float* bih[4] = {(const float*)d_in[10], (const float*)d_in[14],
                         (const float*)d_in[18], (const float*)d_in[22]};
  const float* bhh[4] = {(const float*)d_in[11], (const float*)d_in[15],
                         (const float*)d_in[19], (const float*)d_in[23]};
  const float* nlW = (const float*)d_in[24];
  const float* nlb = (const float*)d_in[25];
  const float* scW = (const float*)d_in[26];
  const float* scb = (const float*)d_in[27];

  char* ws = (char*)d_ws;
  size_t off = 0;
  auto take = [&](size_t bytes)->char*{
    char* p = ws + off;
    off = (off + bytes + 255) & ~(size_t)255;
    return p;
  };
  float*  levs = (float*) take((size_t)NS * TT  * 4);
  float*  seas = (float*) take((size_t)NS * 492 * 4);
  __bf16* inp  = (__bf16*)take((size_t)TPM * NS * 32 * 2);
  __bf16* h1   = (__bf16*)take((size_t)TPM * NS * 64 * 2);
  __bf16* h2   = (__bf16*)take((size_t)TPM * NS * 64 * 2);
  __bf16* cc   = (__bf16*)take((size_t)TPM * NS * 96 * 2);
  __bf16* h3   = (__bf16*)take((size_t)TPM * NS * 64 * 2);
  __bf16* h4   = (__bf16*)take((size_t)TPM * NS * 64 * 2);
  __bf16* hnl  = (__bf16*)take((size_t)TPM * NS * 64 * 2);
  int KpIH[4] = {32, 64, 96, 64};
  int KIH[4]  = {18, 50, 68, 50};
  __bf16* wIH[4]; __bf16* wHH[4]; float* bS[4];
  for (int l = 0; l < 4; ++l){
    wIH[l] = (__bf16*)take((size_t)NGP * KpIH[l] * 2);
    wHH[l] = (__bf16*)take((size_t)NGP * 64 * 2);
    bS[l]  = (float*) take((size_t)NG * 4);
  }
  __bf16* wNL = (__bf16*)take((size_t)64 * 64 * 2);
  __bf16* wSC = (__bf16*)take((size_t)32 * 64 * 2);
  size_t total = off;

  // 0) zero workspace (zero K/time pads; deterministic per launch)
  zero_ws_kernel<<<2048, 256, 0, stream>>>((uint4*)ws, (total + 15) / 16);

  // 1) weight conversions fp32 -> padded bf16, bias sums
  auto convs = [&](const float* s, __bf16* d, int N, int K, int Np, int Kp){
    int tot = Np * Kp;
    conv_weight_kernel<<<(tot + 255) / 256, 256, 0, stream>>>(s, d, N, K, Np, Kp);
  };
  for (int l = 0; l < 4; ++l){
    convs(Wih[l], wIH[l], NG, KIH[l], NGP, KpIH[l]);
    convs(Whh[l], wHH[l], NG, HID, NGP, 64);
    bias_sum_kernel<<<1, 256, 0, stream>>>(bih[l], bhh[l], bS[l], NG);
  }
  convs(nlW, wNL, HID, HID, 64, 64);
  convs(scW, wSC, 18,  HID, 32, 64);

  // 2) exponential smoothing
  smoothing_kernel<<<(NS + 255) / 256, 256, 0, stream>>>(train, ilev, isea,
                                                         iseason, levs, seas);
  // 3) windows + output_batch
  float* outMain  = (float*)d_out;
  float* outBatch = outMain + (size_t)NW * NS * 18;
  build_input_kernel<<<(NW * NS + 255) / 256, 256, 0, stream>>>(
      train, info_cat, levs, seas, inp, outBatch);

  // 4) DRNN1 layer 0: dilation 1,  in=18 (Kp 32), 451 steps x 2048 batch
  lstm_fused_kernel<1><<<NS / 16, 128, 0, stream>>>(
      inp, wIH[0], wHH[0], bS[0], h1, NW, NS);
  // 5) DRNN1 layer 1: dilation 3,  in=50 (Kp 64), 151 steps x 6144 batch
  lstm_fused_kernel<2><<<3 * NS / 16, 128, 0, stream>>>(
      h1, wIH[1], wHH[1], bS[1], h2, 151, 3 * NS);
  // 6) residual concat [input | h2]
  concat_kernel<<<(NW * NS + 255) / 256, 256, 0, stream>>>(inp, h2, cc);
  // 7) DRNN2 layer 0: dilation 6,  in=68 (Kp 96), 76 steps x 12288 batch
  lstm_fused_kernel<3><<<6 * NS / 16, 128, 0, stream>>>(
      cc, wIH[2], wHH[2], bS[2], h3, 76, 6 * NS);
  // 8) DRNN2 layer 1: dilation 12, in=50 (Kp 64), 38 steps x 24576 batch
  lstm_fused_kernel<2><<<12 * NS / 16, 128, 0, stream>>>(
      h3, wIH[3], wHH[3], bS[3], h4, 38, 12 * NS);
  // 9) nonlinear layer: tanh(h4 @ nlW^T + nlb) -> bf16
  gemm_nt4_kernel<4><<<NW * NS / 16, 32, 0, stream>>>(
      h4, 64, wNL, 64, nlb, NW * NS, HID, 64, nullptr, 0, hnl, 64, 1);
  // 10) scoring head -> d_out (f32, [451,2048,18])
  gemm_nt4_kernel<2><<<NW * NS / 16, 32, 0, stream>>>(
      hnl, 64, wSC, 64, scb, NW * NS, 18, 64, outMain, 18, nullptr, 0, 0);
}
